// GraphConvolutionSparse_42391327212274
// MI455X (gfx1250) — matrix operand order repaired and checked by reading, not compiled, not run on hardware
//
#include <hip/hip_runtime.h>
#include <hip/hip_bf16.h>

#define N_NODES 50000
#define IN_DIM  256
#define OUT_DIM 128
#define NNZ_X   800000
#define N_EDGES 1600000

typedef __attribute__((ext_vector_type(16))) __bf16 bf16x16;
typedef __attribute__((ext_vector_type(8)))  float  f32x8;
typedef __attribute__((ext_vector_type(4)))  unsigned int u32x4;
typedef __attribute__((ext_vector_type(8)))  int    i32x8;
typedef __attribute__((ext_vector_type(4)))  int    i32x4;

// ---------------- helpers ----------------
__device__ __forceinline__ unsigned short f2bf(float x) {
    // round-to-nearest-even f32 -> bf16
    unsigned u = __float_as_uint(x);
    unsigned r = u + 0x7FFFu + ((u >> 16) & 1u);
    return (unsigned short)(r >> 16);
}

// ---------------- 1) zero dense bf16 X and row counters ----------------
__global__ void zero_ws_kernel(unsigned* __restrict__ Xd32, unsigned* __restrict__ cnt) {
    long long i = (long long)blockIdx.x * blockDim.x + threadIdx.x;
    const long long nXd = (long long)N_NODES * IN_DIM / 2;      // 6.4M u32 (bf16 pairs)
    if (i < nXd)                 Xd32[i] = 0u;
    else if (i < nXd + N_NODES)  cnt[i - nXd] = 0u;
}

// ---------------- 2) densify sparse X into bf16 via packed-bf16 atomics -----
__global__ void densify_kernel(const int* __restrict__ xr, const int* __restrict__ xc,
                               const float* __restrict__ xv, const float* __restrict__ noise,
                               __hip_bfloat162* __restrict__ Xd2) {
    int i = blockIdx.x * blockDim.x + threadIdx.x;
    if (i >= NNZ_X) return;
    float mask = floorf(1.0f + noise[i]);   // keep_prob = 1.0 -> always 1
    float v = xv[i] * mask;
    int r = xr[i], c = xc[i];
    __hip_bfloat16 bv = __float2bfloat16(v);
    __hip_bfloat16 zb = __float2bfloat16(0.0f);
    __hip_bfloat162 pk;
    if (c & 1) { pk.x = zb; pk.y = bv; } else { pk.x = bv; pk.y = zb; }
    unsafeAtomicAdd(&Xd2[(long long)r * (IN_DIM / 2) + (c >> 1)], pk); // global_atomic_pk_add_bf16
}

// ---------------- 3) histogram adj rows ----------------
__global__ void hist_kernel(const int* __restrict__ ar, unsigned* __restrict__ cnt) {
    int i = blockIdx.x * blockDim.x + threadIdx.x;
    if (i >= N_EDGES) return;
    atomicAdd(&cnt[ar[i]], 1u);
}

// ---------------- 4) W (256x128 f32) -> Wt (128x256 bf16) ----------------
__global__ void wtrans_kernel(const float* __restrict__ W, unsigned short* __restrict__ Wt) {
    int n = blockIdx.x;        // 0..127
    int k = threadIdx.x;       // 0..255
    Wt[n * IN_DIM + k] = f2bf(W[k * OUT_DIM + n]);
}

// ---------------- 5) single-block exclusive scan -> row_start + cursor ------
__global__ __launch_bounds__(1024) void scan_kernel(const unsigned* cnt,
                                                    unsigned* __restrict__ rs,
                                                    unsigned* cursor) {
    __shared__ unsigned s[1024];
    const int t = threadIdx.x;
    unsigned carry = 0;
    for (int base = 0; base < N_NODES; base += 1024) {
        int i = base + t;
        unsigned v = (i < N_NODES) ? cnt[i] : 0u;
        s[t] = v;
        __syncthreads();
        for (int off = 1; off < 1024; off <<= 1) {
            unsigned x = (t >= off) ? s[t - off] : 0u;
            __syncthreads();
            s[t] += x;
            __syncthreads();
        }
        unsigned incl  = s[t];
        unsigned total = s[1023];
        if (i < N_NODES) {
            unsigned excl = carry + incl - v;
            rs[i] = excl;
            cursor[i] = excl;
        }
        carry += total;
        __syncthreads();
    }
    if (t == 0) rs[N_NODES] = carry;
}

// ---------------- 6) scatter edges into CSR order ----------------
__global__ void scatter_kernel(const int* __restrict__ ar, const int* __restrict__ ac,
                               const float* __restrict__ av, unsigned* __restrict__ cursor,
                               int* __restrict__ ccol, float* __restrict__ cval) {
    int i = blockIdx.x * blockDim.x + threadIdx.x;
    if (i >= N_EDGES) return;
    unsigned pos = atomicAdd(&cursor[ar[i]], 1u);
    ccol[pos] = ac[i];
    cval[pos] = av[i];
}

// ---------------- 7) dense GEMM h = Xbf @ W via TDM + WMMA bf16 ----------------
// Block: 80 rows x 128 cols. 8 waves; wave w owns N-tile w, does 5 M-tiles.
// A tile (80 x 256 bf16 = 40KB) staged in LDS by the Tensor Data Mover with
// hardware row padding: row = 512B = 128 DWORDs (pad_interval=6), pad 4 DWORDs
// (pad_amount=3) -> LDS row stride 132 DWORDs = conflict-free fragment reads.
#define AS_STRIDE 132
__global__ __launch_bounds__(256) void gemm_wmma_kernel(const unsigned short* __restrict__ Xbf,
                                                        const unsigned short* __restrict__ Wt,
                                                        float* __restrict__ h) {
    __shared__ unsigned As[80 * AS_STRIDE];   // bf16 pairs, row-major K, padded rows
    const int tid  = threadIdx.x;
    const int row0 = blockIdx.x * 80;         // 625 * 80 = 50000 exactly

#if __has_builtin(__builtin_amdgcn_tensor_load_to_lds)
    if (tid < 32) {   // wave 0 issues one TDM tile copy for the whole block
        unsigned ldsAddr = (unsigned)(unsigned long long)
            (__attribute__((address_space(3))) unsigned*)As;
        unsigned long long ga =
            (unsigned long long)(const void*)(Xbf + (long long)row0 * IN_DIM);
        // D# group 0: count=1 | lds_addr | global_addr | type=2
        u32x4 g0;
        g0[0] = 1u;
        g0[1] = ldsAddr;
        g0[2] = (unsigned)ga;
        g0[3] = (unsigned)(ga >> 32) | (2u << 30);
        // D# group 1: data_size=2B, pad_enable, pad_interval=128dw, pad_amount=4dw,
        //             tensor_dim0=256, tensor_dim1=80, tile_dim0=256, tile_dim1=80,
        //             tensor_dim0_stride=256
        i32x8 g1;
        g1[0] = (int)((1u << 16) | (1u << 20) | (6u << 22) | (3u << 25));
        g1[1] = (int)(256u << 16);   // tensor_dim0[15:0] in bits[31:16]
        g1[2] = (int)(80u  << 16);   // tensor_dim0 hi=0, tensor_dim1[15:0]=80
        g1[3] = (int)(256u << 16);   // tensor_dim1 hi=0, tile_dim0=256
        g1[4] = 80;                  // tile_dim1=80, tile_dim2=0
        g1[5] = 256;                 // tensor_dim0_stride[31:0]
        g1[6] = 0;
        g1[7] = 0;
        i32x4 gz = (i32x4)0;
#if defined(__clang_major__) && (__clang_major__ >= 23)
        __builtin_amdgcn_tensor_load_to_lds(g0, g1, gz, gz, (i32x8)0, 0);
#else
        __builtin_amdgcn_tensor_load_to_lds(g0, g1, gz, gz, 0);
#endif
        __builtin_amdgcn_s_wait_tensorcnt(0);
    }
    __syncthreads();
#else
    // fallback: cooperative bf16 copy (no conversion needed, Xbf already bf16)
    const unsigned* X32 = (const unsigned*)(Xbf + (long long)row0 * IN_DIM);
    for (int idx = tid; idx < 80 * 128; idx += 256) {
        int r = idx >> 7, k2 = idx & 127;
        As[r * AS_STRIDE + k2] = X32[r * 128 + k2];
    }
    __syncthreads();
#endif

    const int wave  = tid >> 5;     // 0..7 -> N tile
    const int lane  = tid & 31;
    const int lhalf = lane >> 4;    // 0/1
    const int l16   = lane & 15;
    const int nbase = wave * 16;

    union Frag { struct { uint4 lo, hi; } u; bf16x16 v; };

    // B fragments from global bf16 Wt[128][256] (L2-hot).
    // Lane layout: column N = l16 (+nbase), K = kt*32 + lhalf*16 .. +15 contiguous.
    Frag b[8];
    const uint4* Wt4 = (const uint4*)(Wt + (long long)(nbase + l16) * IN_DIM);
    #pragma unroll
    for (int kt = 0; kt < 8; ++kt) {
        int koff16 = kt * 32 + lhalf * 16;
        b[kt].u.lo = Wt4[koff16 >> 3];
        b[kt].u.hi = Wt4[(koff16 >> 3) + 1];
    }

    f32x8 acc[5] = {};

    #pragma unroll
    for (int kt = 0; kt < 8; ++kt) {
        #pragma unroll
        for (int mt = 0; mt < 5; ++mt) {
            // A lane layout: row M = l16 (+tile), K = kt*32 + lhalf*16 .. +15 contiguous
            Frag a;
            int arow = mt * 16 + l16;
            int kd   = (kt * 32 + lhalf * 16) >> 1;   // dword offset (multiple of 4)
            const uint4* ap = (const uint4*)&As[arow * AS_STRIDE + kd];
            a.u.lo = ap[0];   // K kb .. kb+7
            a.u.hi = ap[1];   // K kb+8 .. kb+15
            acc[mt] = __builtin_amdgcn_wmma_f32_16x16x32_bf16(
                false, a.v, false, b[kt].v, (short)0, acc[mt], false, false);
        }
    }

    // C/D layout: N = l16, M = lhalf*8 + r (VGPR r)
    const int mofs = lhalf * 8;
    #pragma unroll
    for (int mt = 0; mt < 5; ++mt) {
        int rbase = row0 + mt * 16 + mofs;
        #pragma unroll
        for (int r = 0; r < 8; ++r)
            h[(long long)(rbase + r) * OUT_DIM + nbase + l16] = acc[mt][r];
    }
}

// ---------------- 8) out = relu(A @ h), one wave32 per row, CSR gather -------
__global__ __launch_bounds__(256) void spmm_relu_kernel(const unsigned* __restrict__ rs,
                                                        const int* __restrict__ ccol,
                                                        const float* __restrict__ cval,
                                                        const float* __restrict__ h,
                                                        float* __restrict__ out) {
    int row  = blockIdx.x * 8 + (threadIdx.x >> 5);   // 6250 * 8 = 50000
    int lane = threadIdx.x & 31;
    if (row >= N_NODES) return;
    unsigned s = rs[row], e = rs[row + 1];
    float4 acc = {0.f, 0.f, 0.f, 0.f};
    const float4* h4 = (const float4*)h;              // 32 float4 per node row
    for (unsigned i = s; i < e; ++i) {
        int c   = ccol[i];                            // uniform per wave
        float w = cval[i];
        float4 v = h4[(long long)c * 32 + lane];      // 512B L2-resident gather
        acc.x += w * v.x; acc.y += w * v.y; acc.z += w * v.z; acc.w += w * v.w;
    }
    float4 r;
    r.x = fmaxf(acc.x, 0.f); r.y = fmaxf(acc.y, 0.f);
    r.z = fmaxf(acc.z, 0.f); r.w = fmaxf(acc.w, 0.f);
    ((float4*)out)[(long long)row * 32 + lane] = r;
}

// ---------------- launcher ----------------
extern "C" void kernel_launch(void* const* d_in, const int* in_sizes, int n_in,
                              void* d_out, int out_size, void* d_ws, size_t ws_size,
                              hipStream_t stream) {
    const int*   x_rows   = (const int*)d_in[0];
    const int*   x_cols   = (const int*)d_in[1];
    const float* x_vals   = (const float*)d_in[2];
    const float* noise    = (const float*)d_in[3];
    const int*   adj_rows = (const int*)d_in[4];
    const int*   adj_cols = (const int*)d_in[5];
    const float* adj_vals = (const float*)d_in[6];
    const float* W        = (const float*)d_in[7];
    float*       out      = (float*)d_out;

    // workspace carve-up (units: u32 words from base)
    unsigned* ws32 = (unsigned*)d_ws;
    unsigned short* Xbf  = (unsigned short*)(ws32);              //  12,800,000 bf16 (6.4M u32)
    float*          h    = (float*)(ws32 + 6400000);             //   6,400,000 f32
    unsigned*       rs   = (unsigned*)(ws32 + 12800000);         //      50,001 u32
    unsigned*       cnt  = (unsigned*)(ws32 + 12850016);         //      50,000 u32 (also cursor)
    int*            ccol = (int*)(ws32 + 12900016);              //   1,600,000 i32
    float*          cval = (float*)(ws32 + 14500016);            //   1,600,000 f32
    unsigned short* Wt   = (unsigned short*)(ws32 + 16100016);   //      32,768 bf16

    const long long zeroWords = (long long)N_NODES * IN_DIM / 2 + N_NODES; // 6,450,000
    zero_ws_kernel<<<(int)((zeroWords + 1023) / 1024), 1024, 0, stream>>>((unsigned*)Xbf, cnt);
    densify_kernel<<<(NNZ_X + 255) / 256, 256, 0, stream>>>(x_rows, x_cols, x_vals, noise,
                                                            (__hip_bfloat162*)Xbf);
    hist_kernel<<<(N_EDGES + 255) / 256, 256, 0, stream>>>(adj_rows, cnt);
    wtrans_kernel<<<OUT_DIM, IN_DIM, 0, stream>>>(W, Wt);
    scan_kernel<<<1, 1024, 0, stream>>>(cnt, rs, cnt);
    scatter_kernel<<<(N_EDGES + 255) / 256, 256, 0, stream>>>(adj_rows, adj_cols, adj_vals,
                                                              cnt, ccol, cval);
    gemm_wmma_kernel<<<625, 256, 0, stream>>>(Xbf, Wt, h);
    spmm_relu_kernel<<<6250, 256, 0, stream>>>(rs, ccol, cval, h, out);
}
